// TrellisMoEMLP_50723563766121
// MI455X (gfx1250) — compile-verified, matrix operand order checked by reading, not run on hardware
//
#include <hip/hip_runtime.h>

// ---------------- problem constants ----------------
#define HID    2048
#define INTER  512
#define SINTER 1024
#define NEXP   64
#define TOPK   8
#define NTOK   2048

// ---------------- tiling ----------------
#define BM  32     // token rows per block (all waves cover all rows)
#define BN  256    // output cols per block (8 waves x 32 cols)
#define BK  32     // K step (== WMMA K for f16)
#define LDT 48     // padded LDS row stride in halves (96B -> 32B-aligned frags)

typedef __attribute__((ext_vector_type(16))) _Float16 v16h;
typedef __attribute__((ext_vector_type(8)))  float    v8f;

// =====================================================================
// helpers
// =====================================================================
// Load a 16-row x 32-K A-fragment slice (16 contiguous fp32) -> f16 frag.
__device__ __forceinline__ v16h load_a_frag_f32(const float* __restrict__ p) {
  v16h a;
  #pragma unroll
  for (int q = 0; q < 4; ++q) {
    float4 v = ((const float4*)p)[q];
    a[4 * q + 0] = (_Float16)v.x; a[4 * q + 1] = (_Float16)v.y;
    a[4 * q + 2] = (_Float16)v.z; a[4 * q + 3] = (_Float16)v.w;
  }
  return a;
}

// Stage one fp32 weight tile [BK x BN] transposed into LDS as f16.
// thread tid owns column n=tid; LD is compile-time so every global
// load uses an immediate offset off one base address (no xcnt chain).
template<int LD>
__device__ __forceinline__ void stage_w(_Float16* __restrict__ Wt,
                                        const float* __restrict__ wp,
                                        int tid) {
  #pragma unroll
  for (int k = 0; k < BK; ++k)
    Wt[tid * LDT + k] = (_Float16)wp[(size_t)k * LD];
}

// =====================================================================
// Router: one block (64 threads) per token. Top-8 of logits + softmax
// over the selected logits == softmax -> top-k -> renormalize.
// =====================================================================
__global__ __launch_bounds__(64) void router_kernel(
    const float* __restrict__ x, const float* __restrict__ rwm,
    int* __restrict__ sel, float* __restrict__ rwts)
{
  const int t = blockIdx.x;
  const int e = threadIdx.x;
  __shared__ float logits[NEXP];

  const float4* xr = (const float4*)(x + (size_t)t * HID);
  const float4* wr = (const float4*)(rwm + (size_t)e * HID);
  float acc = 0.f;
  #pragma unroll 4
  for (int i = 0; i < HID / 4; ++i) {
    float4 a = xr[i], b = wr[i];
    acc += a.x * b.x + a.y * b.y + a.z * b.z + a.w * b.w;
  }
  logits[e] = acc;
  __syncthreads();

  if (e == 0) {
    float l[NEXP];
    #pragma unroll
    for (int i = 0; i < NEXP; ++i) l[i] = logits[i];
    int   ids[TOPK];
    float vals[TOPK];
    for (int k = 0; k < TOPK; ++k) {
      int bi = 0; float bv = -1e30f;
      for (int i = 0; i < NEXP; ++i) { if (l[i] > bv) { bv = l[i]; bi = i; } }
      ids[k] = bi; vals[k] = bv; l[bi] = -1e30f;
    }
    const float m = vals[0];
    float p[TOPK], s = 0.f;
    #pragma unroll
    for (int k = 0; k < TOPK; ++k) { p[k] = __expf(vals[k] - m); s += p[k]; }
    const float inv = 1.f / s;
    #pragma unroll
    for (int k = 0; k < TOPK; ++k) {
      sel [t * TOPK + k] = ids[k];
      rwts[t * TOPK + k] = p[k] * inv;
    }
  }
}

// =====================================================================
// Bucket the 16384 (token,k) slots into per-expert lists.
// =====================================================================
__global__ void build_lists_kernel(const int* __restrict__ sel,
                                   int* __restrict__ counts,
                                   int* __restrict__ toklist)
{
  int s = blockIdx.x * blockDim.x + threadIdx.x;
  if (s < NTOK * TOPK) {
    int e   = sel[s];
    int pos = atomicAdd(&counts[e], 1);
    toklist[e * NTOK + pos] = s;  // slot encodes token = s>>3
  }
}

// =====================================================================
// Grouped gate+up GEMM with fused SiLU.
// Block = (expert, 32-token tile, 256-col chunk); 8 waves, each wave
// owns the full 32 rows x 32 cols -> 8 WMMAs per K-step off only
// 4 LDS B-fragments. A-fragments come straight from global x (L2-
// resident) with in-register f32->f16 conversion. Weight tiles are
// double-buffered in dynamic LDS: one barrier per K-step, streaming
// overlaps the WMMA chain; K+2 tile prefetched (global_prefetch_b8).
// MOE=false -> shared expert (identity token list).
// =====================================================================
template<int LD, bool MOE>
__global__ __launch_bounds__(256) void gateup_kernel(
    const float* __restrict__ x,
    const int* __restrict__ toklist, const int* __restrict__ counts,
    const float* __restrict__ wg_all, const float* __restrict__ wu_all,
    _Float16* __restrict__ hout)
{
  const int e     = blockIdx.x;
  const int mbase = blockIdx.y * BM;
  const int nbase = blockIdx.z * BN;

  int ne;
  if (MOE) { ne = counts[e]; if (mbase >= ne) return; }
  else     { ne = NTOK; }

  const float* wg = wg_all + (size_t)e * HID * LD;
  const float* wu = wu_all + (size_t)e * HID * LD;

  // dynamic LDS: [buf][Gt BN*LDT | Ut BN*LDT] halves, double buffered
  extern __shared__ __align__(32) _Float16 smem_h[];
  const int BUFH = 2 * BN * LDT;  // halves per buffer

  __shared__ int rowH[BM];
  __shared__ int rowT[BM];

  const int tid = threadIdx.x;
  if (tid < BM) {
    int r = mbase + tid, hrow = -1, tok = 0;
    if (r < ne) {
      if (MOE) { int slot = toklist[e * NTOK + r]; hrow = slot; tok = slot >> 3; }
      else     { hrow = r; tok = r; }
    }
    rowH[tid] = hrow; rowT[tid] = tok;
  }

  const int wave  = tid >> 5, lane = tid & 31;
  const int halfi = lane >> 4, idx = lane & 15;
  const int wcol  = wave * 32;             // this wave's col base in BN

  const float* gp0 = wg + nbase + tid;     // + k*LD walks rows
  const float* up0 = wu + nbase + tid;

  v8f accG[2][2] = {}, accU[2][2] = {};

  // prologue: stage K-step 0 into buffer 0
  stage_w<LD>(smem_h,            gp0, tid);
  stage_w<LD>(smem_h + BN * LDT, up0, tid);
  __syncthreads();

  const int nsteps = HID / BK;
  for (int s = 0; s < nsteps; ++s) {
    const int buf = s & 1;
    const int k0  = s * BK;

    // stage next tile into the other buffer (overlaps WMMA below)
    if (s + 1 < nsteps) {
      const size_t koff = (size_t)(k0 + BK) * LD;
      stage_w<LD>(smem_h + (buf ^ 1) * BUFH,            gp0 + koff, tid);
      stage_w<LD>(smem_h + (buf ^ 1) * BUFH + BN * LDT, up0 + koff, tid);
    }
    // prefetch the K+2 tile toward L2/L0
    if (s + 2 < nsteps) {
      const size_t poff = (size_t)(k0 + 2 * BK) * LD;
      __builtin_prefetch(gp0 + poff, 0, 1);
      __builtin_prefetch(up0 + poff, 0, 1);
    }

    // ---- gather all fragments, then back-to-back WMMAs ----
    const _Float16* Gt = smem_h + buf * BUFH;
    const _Float16* Ut = Gt + BN * LDT;

    v16h a0 = load_a_frag_f32(x + (size_t)rowT[idx]      * HID + k0 + halfi * 16);
    v16h a1 = load_a_frag_f32(x + (size_t)rowT[16 + idx] * HID + k0 + halfi * 16);
    v16h bg0 = *(const v16h*)&Gt[(wcol +  0 + idx) * LDT + halfi * 16];
    v16h bg1 = *(const v16h*)&Gt[(wcol + 16 + idx) * LDT + halfi * 16];
    v16h bu0 = *(const v16h*)&Ut[(wcol +  0 + idx) * LDT + halfi * 16];
    v16h bu1 = *(const v16h*)&Ut[(wcol + 16 + idx) * LDT + halfi * 16];

    accG[0][0] = __builtin_amdgcn_wmma_f32_16x16x32_f16(false, a0, false, bg0, (short)0, accG[0][0], false, false);
    accG[0][1] = __builtin_amdgcn_wmma_f32_16x16x32_f16(false, a0, false, bg1, (short)0, accG[0][1], false, false);
    accG[1][0] = __builtin_amdgcn_wmma_f32_16x16x32_f16(false, a1, false, bg0, (short)0, accG[1][0], false, false);
    accG[1][1] = __builtin_amdgcn_wmma_f32_16x16x32_f16(false, a1, false, bg1, (short)0, accG[1][1], false, false);
    accU[0][0] = __builtin_amdgcn_wmma_f32_16x16x32_f16(false, a0, false, bu0, (short)0, accU[0][0], false, false);
    accU[0][1] = __builtin_amdgcn_wmma_f32_16x16x32_f16(false, a0, false, bu1, (short)0, accU[0][1], false, false);
    accU[1][0] = __builtin_amdgcn_wmma_f32_16x16x32_f16(false, a1, false, bu0, (short)0, accU[1][0], false, false);
    accU[1][1] = __builtin_amdgcn_wmma_f32_16x16x32_f16(false, a1, false, bu1, (short)0, accU[1][1], false, false);

    __syncthreads();
  }

  // --- fused SiLU(gate) * up epilogue, store f16 h ---
  #pragma unroll
  for (int mi = 0; mi < 2; ++mi) {
    #pragma unroll
    for (int ni = 0; ni < 2; ++ni) {
      #pragma unroll
      for (int r = 0; r < 8; ++r) {
        int m    = mi * 16 + halfi * 8 + r;
        int hrow = rowH[m];
        if (hrow >= 0) {
          float g = accG[mi][ni][r], u = accU[mi][ni][r];
          float hv = (g / (1.f + __expf(-g))) * u;   // silu(g)*u
          hout[(size_t)hrow * LD + nbase + wcol + ni * 16 + idx] = (_Float16)hv;
        }
      }
    }
  }
}

// =====================================================================
// Down projection. A-fragments loaded directly from the f16 h buffer
// (fragment-contiguous, 32B global loads); Wd tiles double-buffered
// in dynamic LDS. MOE=false (shared expert): direct store, covers all
// of out. MOE=true: scale by router weight, atomic f32 add.
// =====================================================================
template<int KD, bool MOE>
__global__ __launch_bounds__(256) void down_kernel(
    const _Float16* __restrict__ hbuf,
    const int* __restrict__ toklist, const int* __restrict__ counts,
    const float* __restrict__ rwts,
    const float* __restrict__ wd_all,
    float* __restrict__ out)
{
  const int e     = blockIdx.x;
  const int mbase = blockIdx.y * BM;
  const int nbase = blockIdx.z * BN;

  int ne;
  if (MOE) { ne = counts[e]; if (mbase >= ne) return; }
  else     { ne = NTOK; }

  const float* wd = wd_all + (size_t)e * KD * HID;

  extern __shared__ __align__(32) _Float16 smem_h[];
  const int BUFH = BN * LDT;  // halves per buffer (single matrix)

  __shared__ int   rowH[BM];
  __shared__ int   rowT[BM];
  __shared__ float rowW[BM];

  const int tid = threadIdx.x;
  if (tid < BM) {
    int r = mbase + tid, hrow = 0, tok = -1;
    float w = 1.f;
    if (r < ne) {
      if (MOE) {
        int slot = toklist[e * NTOK + r];
        hrow = slot; tok = slot >> 3; w = rwts[slot];
      } else { hrow = r; tok = r; }
    }
    rowH[tid] = hrow; rowT[tid] = tok; rowW[tid] = w;
  }

  const int wave  = tid >> 5, lane = tid & 31;
  const int halfi = lane >> 4, idx = lane & 15;
  const int wcol  = wave * 32;

  const float* dp0 = wd + nbase + tid;

  v8f accD[2][2] = {};

  stage_w<HID>(smem_h, dp0, tid);
  __syncthreads();

  const int nsteps = KD / BK;
  for (int s = 0; s < nsteps; ++s) {
    const int buf = s & 1;
    const int k0  = s * BK;

    if (s + 1 < nsteps)
      stage_w<HID>(smem_h + (buf ^ 1) * BUFH, dp0 + (size_t)(k0 + BK) * HID, tid);
    if (s + 2 < nsteps)
      __builtin_prefetch(dp0 + (size_t)(k0 + 2 * BK) * HID, 0, 1);

    const _Float16* Dt = smem_h + buf * BUFH;

    v16h a0 = *(const v16h*)(hbuf + (size_t)rowH[idx]      * KD + k0 + halfi * 16);
    v16h a1 = *(const v16h*)(hbuf + (size_t)rowH[16 + idx] * KD + k0 + halfi * 16);
    v16h b0 = *(const v16h*)&Dt[(wcol +  0 + idx) * LDT + halfi * 16];
    v16h b1 = *(const v16h*)&Dt[(wcol + 16 + idx) * LDT + halfi * 16];

    accD[0][0] = __builtin_amdgcn_wmma_f32_16x16x32_f16(false, a0, false, b0, (short)0, accD[0][0], false, false);
    accD[0][1] = __builtin_amdgcn_wmma_f32_16x16x32_f16(false, a0, false, b1, (short)0, accD[0][1], false, false);
    accD[1][0] = __builtin_amdgcn_wmma_f32_16x16x32_f16(false, a1, false, b0, (short)0, accD[1][0], false, false);
    accD[1][1] = __builtin_amdgcn_wmma_f32_16x16x32_f16(false, a1, false, b1, (short)0, accD[1][1], false, false);

    __syncthreads();
  }

  #pragma unroll
  for (int mi = 0; mi < 2; ++mi) {
    #pragma unroll
    for (int ni = 0; ni < 2; ++ni) {
      #pragma unroll
      for (int r = 0; r < 8; ++r) {
        int m   = mi * 16 + halfi * 8 + r;
        int tok = rowT[m];
        int col = nbase + wcol + ni * 16 + idx;
        if (MOE) {
          if (tok >= 0)
            atomicAdd(&out[(size_t)tok * HID + col], rowW[m] * accD[mi][ni][r]);
        } else {
          out[(size_t)tok * HID + col] = accD[mi][ni][r];
        }
      }
    }
  }
}

// =====================================================================
extern "C" void kernel_launch(void* const* d_in, const int* in_sizes, int n_in,
                              void* d_out, int out_size, void* d_ws, size_t ws_size,
                              hipStream_t stream)
{
  const float* x        = (const float*)d_in[0];
  const float* router_w = (const float*)d_in[1];
  const float* wg       = (const float*)d_in[2];
  const float* wu       = (const float*)d_in[3];
  const float* wd       = (const float*)d_in[4];
  const float* sg       = (const float*)d_in[5];
  const float* su       = (const float*)d_in[6];
  const float* sd       = (const float*)d_in[7];
  float* out = (float*)d_out;

  // workspace layout
  char* ws = (char*)d_ws;
  size_t off = 0;
  int*      sel     = (int*)(ws + off);        off += (size_t)NTOK * TOPK * 4;   // 64KB
  float*    rwts    = (float*)(ws + off);      off += (size_t)NTOK * TOPK * 4;   // 64KB
  int*      counts  = (int*)(ws + off);        off += 1024;                      // 64 ints, padded
  int*      toklist = (int*)(ws + off);        off += (size_t)NEXP * NTOK * 4;   // 512KB
  _Float16* hmoe    = (_Float16*)(ws + off);   off += (size_t)NTOK * TOPK * INTER * 2; // 16MB
  _Float16* hsh     = (_Float16*)(ws + off);   off += (size_t)NTOK * SINTER * 2;       // 4MB

  const size_t lds_gateup = (size_t)2 * 2 * BN * LDT * sizeof(_Float16); // 96KB
  const size_t lds_down   = (size_t)2 * BN * LDT * sizeof(_Float16);     // 48KB

  // 1) router
  router_kernel<<<NTOK, 64, 0, stream>>>(x, router_w, sel, rwts);

  // 2) per-expert token lists
  hipMemsetAsync(counts, 0, NEXP * sizeof(int), stream);
  build_lists_kernel<<<(NTOK * TOPK + 255) / 256, 256, 0, stream>>>(sel, counts, toklist);

  // 3) grouped gate+up (MoE experts), f16 WMMA
  gateup_kernel<INTER, true><<<dim3(NEXP, NTOK / BM, INTER / BN), 256, lds_gateup, stream>>>(
      x, toklist, counts, wg, wu, hmoe);

  // 4) shared-expert gate+up
  gateup_kernel<SINTER, false><<<dim3(1, NTOK / BM, SINTER / BN), 256, lds_gateup, stream>>>(
      x, nullptr, nullptr, sg, su, hsh);

  // 5) shared-expert down: writes every element of out (no memset needed)
  down_kernel<SINTER, false><<<dim3(1, NTOK / BM, HID / BN), 256, lds_down, stream>>>(
      hsh, nullptr, nullptr, nullptr, sd, out);

  // 6) grouped MoE down: scaled atomic-add into out
  down_kernel<INTER, true><<<dim3(NEXP, NTOK / BM, HID / BN), 256, lds_down, stream>>>(
      hmoe, toklist, counts, rwts, wd, out);
}
